// MultiDimensinalGRUCell_4956392260353
// MI455X (gfx1250) — matrix operand order repaired and checked by reading, not compiled, not run on hardware
//
#include <hip/hip_runtime.h>

typedef __bf16 bf16_t;
typedef __attribute__((ext_vector_type(16))) __bf16 v16bf;
typedef __attribute__((ext_vector_type(8)))  __bf16 v8bf;
typedef __attribute__((ext_vector_type(8)))  float   v8f;
typedef __attribute__((ext_vector_type(4)))  int     v4i;

#define BQ    8192
#define DIMK  4096      // K for both GEMMs
#define N1    7168      // 7 gates * U
#define UDIM  1024

#define TM 128
#define TN 256
#define TK 32
#define ABUF (TM * TK)  // LDS elements per A buffer
#define BBUF (TK * TN)  // LDS elements per B buffer

// CDNA5 async global->LDS copies (ASYNCcnt path), if the toolchain exposes them.
#if defined(__has_builtin)
#if __has_builtin(__builtin_amdgcn_global_load_async_to_lds_b128) && \
    __has_builtin(__builtin_amdgcn_s_wait_asynccnt)
#define USE_ASYNC_LDS 1
#endif
#endif
#ifndef USE_ASYNC_LDS
#define USE_ASYNC_LDS 0
#endif

#if USE_ASYNC_LDS
// builtin signature (from compiler diagnostic): (int4 as1*, int4 as3*, imm, imm)
typedef __attribute__((address_space(1))) v4i gv4i_t;
typedef __attribute__((address_space(3))) v4i lv4i_t;
static __device__ __forceinline__ void glds_cp128(bf16_t* lds, const bf16_t* g) {
  // generic global VA == as1 VA; generic LDS pointer low 32 bits == LDS offset
  __builtin_amdgcn_global_load_async_to_lds_b128(
      (gv4i_t*)(unsigned long long)g,
      (lv4i_t*)(unsigned)(unsigned long long)lds,
      0, 0);
}
#endif

static __device__ __forceinline__ bf16_t f2bf(float f) {
  union { float f; unsigned u; } v; v.f = f;
  unsigned r = (v.u + 0x7FFFu + ((v.u >> 16) & 1u)) >> 16;   // RNE
  unsigned short h = (unsigned short)r;
  bf16_t b;
  __builtin_memcpy(&b, &h, sizeof(b));
  return b;
}

static __device__ __forceinline__ v16bf cat8(v8bf lo, v8bf hi) {
  return __builtin_shufflevector(lo, hi, 0,1,2,3,4,5,6,7,8,9,10,11,12,13,14,15);
}

// One K-step of WMMAs for this wave from an LDS buffer pair.
static __device__ __forceinline__ void wmma_step(
    const bf16_t* As, const bf16_t* Bs, int mBase, int nBase,
    int r16, int khalf, int lane, v8f acc[4][4])
{
  v16bf afrag[4], bfrag[4];
  // A fragment (16x32): lane l<16 -> row l, K {0..7,16..23}; lane>=16 -> K {8..15,24..31}
  #pragma unroll
  for (int i = 0; i < 4; ++i) {
    const bf16_t* p = As + (mBase + 16 * i + r16) * TK + khalf;
    afrag[i] = cat8(*(const v8bf*)p, *(const v8bf*)(p + 16));
  }
  // B fragment (32x16): lane l -> col l%16, K {0..7,16..23} (l<16) / {8..15,24..31} (l>=16)
  #pragma unroll
  for (int j = 0; j < 4; ++j) {
    const bf16_t* p = Bs + khalf * TN + nBase + 16 * j + r16;
    #pragma unroll
    for (int s = 0; s < 8; ++s) {
      bfrag[j][s]     = p[s * TN];
      bfrag[j][s + 8] = p[(s + 16) * TN];
    }
  }
  #pragma unroll
  for (int i = 0; i < 4; ++i)
    #pragma unroll
    for (int j = 0; j < 4; ++j)
      acc[i][j] = __builtin_amdgcn_wmma_f32_16x16x32_bf16(
          false, afrag[i], false, bfrag[j], (short)0, acc[i][j], false, false);
}

// ---------------------------------------------------------------------------
// Double-buffered WMMA mainloop: C(TMxTN) += A(rowTile.., K) * B(K, colTile..)
// 8 waves (2x4), each wave owns a 64x64 tile = 4x4 accumulators of 16x16.
// One barrier per K-step; tile k+1 is copied (async if available) into the
// alternate LDS buffer while the WMMAs run on tile k.
// ---------------------------------------------------------------------------
static __device__ __forceinline__ void gemm_mainloop(
    const bf16_t* __restrict__ A, const bf16_t* __restrict__ B, int N,
    int rowTile, int colTile, bf16_t* As, bf16_t* Bs, v8f acc[4][4])
{
  const int tid   = threadIdx.x;
  const int lane  = tid & 31;
  const int wave  = tid >> 5;
  const int mBase = (wave >> 2) * 64;       // 0 / 64
  const int nBase = (wave & 3) * 64;        // 0 / 64 / 128 / 192
  const int r16   = lane & 15;
  const int khalf = (lane >> 4) * 8;

  // staging geometry: A tile 128x32 (2 x 16B per thread), B tile 32x256 (4 x 16B)
  const int aRow = tid >> 2, aCol = (tid & 3) << 3;
  const int bRow = tid >> 5, bCol = (tid & 31) << 3;

  const bf16_t* ag0 = A + (size_t)(rowTile + aRow) * DIMK + aCol;
  const bf16_t* ag1 = ag0 + (size_t)64 * DIMK;
  const bf16_t* bg0 = B + (size_t)bRow * N + colTile + bCol;
  const bf16_t* bg1 = bg0 + (size_t)8  * N;
  const bf16_t* bg2 = bg0 + (size_t)16 * N;
  const bf16_t* bg3 = bg0 + (size_t)24 * N;
  const size_t bStep = (size_t)TK * N;

  bf16_t* lA0 = As + aRow * TK + aCol;
  bf16_t* lA1 = lA0 + 64 * TK;
  bf16_t* lB0 = Bs + bRow * TN + bCol;
  bf16_t* lB1 = lB0 + 8  * TN;
  bf16_t* lB2 = lB0 + 16 * TN;
  bf16_t* lB3 = lB0 + 24 * TN;

  v8f zero = {};
  #pragma unroll
  for (int i = 0; i < 4; ++i)
    #pragma unroll
    for (int j = 0; j < 4; ++j)
      acc[i][j] = zero;

#if USE_ASYNC_LDS
  // prologue: async-stage tile 0 into buffer 0
  glds_cp128(lA0, ag0);
  glds_cp128(lA1, ag1);
  glds_cp128(lB0, bg0);
  glds_cp128(lB1, bg1);
  glds_cp128(lB2, bg2);
  glds_cp128(lB3, bg3);
  ag0 += TK; ag1 += TK;
  bg0 += bStep; bg1 += bStep; bg2 += bStep; bg3 += bStep;
  __builtin_amdgcn_s_wait_asynccnt(0);
  __syncthreads();

  int buf = 0;
  for (int k0 = TK; k0 < DIMK; k0 += TK) {
    const int nb = buf ^ 1;
    // async copies straight into the retired buffer; run behind the WMMAs
    glds_cp128(lA0 + nb * ABUF, ag0);
    glds_cp128(lA1 + nb * ABUF, ag1);
    glds_cp128(lB0 + nb * BBUF, bg0);
    glds_cp128(lB1 + nb * BBUF, bg1);
    glds_cp128(lB2 + nb * BBUF, bg2);
    glds_cp128(lB3 + nb * BBUF, bg3);
    ag0 += TK; ag1 += TK;
    bg0 += bStep; bg1 += bStep; bg2 += bStep; bg3 += bStep;

    wmma_step(As + buf * ABUF, Bs + buf * BBUF, mBase, nBase, r16, khalf, lane, acc);

    __builtin_amdgcn_s_wait_asynccnt(0);
    __syncthreads();
    buf = nb;
  }
  wmma_step(As + buf * ABUF, Bs + buf * BBUF, mBase, nBase, r16, khalf, lane, acc);
#else
  // prologue: stage tile 0 into buffer 0 via VGPRs
  *(uint4*)lA0 = *(const uint4*)ag0;
  *(uint4*)lA1 = *(const uint4*)ag1;
  *(uint4*)lB0 = *(const uint4*)bg0;
  *(uint4*)lB1 = *(const uint4*)bg1;
  *(uint4*)lB2 = *(const uint4*)bg2;
  *(uint4*)lB3 = *(const uint4*)bg3;
  ag0 += TK; ag1 += TK;
  bg0 += bStep; bg1 += bStep; bg2 += bStep; bg3 += bStep;
  __syncthreads();

  int buf = 0;
  for (int k0 = TK; k0 < DIMK; k0 += TK) {
    uint4 a0 = *(const uint4*)ag0, a1 = *(const uint4*)ag1;
    uint4 b0 = *(const uint4*)bg0, b1 = *(const uint4*)bg1;
    uint4 b2 = *(const uint4*)bg2, b3 = *(const uint4*)bg3;
    ag0 += TK; ag1 += TK;
    bg0 += bStep; bg1 += bStep; bg2 += bStep; bg3 += bStep;
    if (k0 + TK < DIMK) {
      __builtin_prefetch(ag0, 0, 1);
      __builtin_prefetch(bg0, 0, 1);
    }

    wmma_step(As + buf * ABUF, Bs + buf * BBUF, mBase, nBase, r16, khalf, lane, acc);

    const int nb = buf ^ 1;
    *(uint4*)(lA0 + nb * ABUF) = a0;
    *(uint4*)(lA1 + nb * ABUF) = a1;
    *(uint4*)(lB0 + nb * BBUF) = b0;
    *(uint4*)(lB1 + nb * BBUF) = b1;
    *(uint4*)(lB2 + nb * BBUF) = b2;
    *(uint4*)(lB3 + nb * BBUF) = b3;
    __syncthreads();
    buf = nb;
  }
  wmma_step(As + buf * ABUF, Bs + buf * BBUF, mBase, nBase, r16, khalf, lane, acc);
#endif
}

// ---------------------------------------------------------------------------
// GEMM1: g = q @ Wb + bias.  Fused epilogue:
//   cols [0,3072):  rs = sigmoid(g) * state  -> bf16 into A2 (GEMM2 A operand)
//   cols [3072,7168): raw logits fp32 -> gz (B x 4096) for softmax
// ---------------------------------------------------------------------------
__global__ __launch_bounds__(256) void gemm1_kernel(
    const bf16_t* __restrict__ q, const bf16_t* __restrict__ Wb,
    const float* __restrict__ bias,
    const float* __restrict__ left, const float* __restrict__ top,
    const float* __restrict__ diag,
    bf16_t* __restrict__ A2, float* __restrict__ gz)
{
  __shared__ bf16_t As[2 * ABUF];
  __shared__ bf16_t Bs[2 * BBUF];

  const int tilesN  = N1 / TN;                    // 28
  const int rowTile = (blockIdx.x / tilesN) * TM;
  const int colTile = (blockIdx.x % tilesN) * TN;

  v8f acc[4][4];
  gemm_mainloop(q, Wb, N1, rowTile, colTile, As, Bs, acc);

  const int lane  = threadIdx.x & 31;
  const int wave  = threadIdx.x >> 5;
  const int mBase = (wave >> 2) * 64;
  const int nBase = (wave & 3) * 64;
  const bool isReset = (colTile < 3 * UDIM);      // TN divides 3072 -> uniform

  #pragma unroll
  for (int i = 0; i < 4; ++i) {
    #pragma unroll
    for (int j = 0; j < 4; ++j) {
      const int col = colTile + nBase + 16 * j + (lane & 15);
      #pragma unroll
      for (int r = 0; r < 8; ++r) {
        const int row = rowTile + mBase + 16 * i + r + 8 * (lane >> 4);
        float g = acc[i][j][r] + bias[col];
        if (isReset) {
          const int piece = col >> 10;
          const float* sp = (piece == 0) ? left : (piece == 1) ? top : diag;
          const float s  = sp[(size_t)row * UDIM + (col & (UDIM - 1))];
          const float rv = 1.0f / (1.0f + __expf(-g));
          A2[(size_t)row * DIMK + col] = f2bf(rv * s);
        } else {
          gz[(size_t)row * DIMK + (col - 3 * UDIM)] = g;
        }
      }
    }
  }
}

// ---------------------------------------------------------------------------
// Softmax across the 4 z gates, in place on gz (B x 4096 = B x [zi|zl|zt|zd])
// ---------------------------------------------------------------------------
__global__ __launch_bounds__(256) void softmax4_kernel(float* __restrict__ gz)
{
  size_t t = (size_t)blockIdx.x * blockDim.x + threadIdx.x;
  if (t >= (size_t)BQ * UDIM) return;
  float* p = gz + (t >> 10) * (size_t)DIMK + (t & (UDIM - 1));
  float a = p[0], b = p[UDIM], c = p[2 * UDIM], d = p[3 * UDIM];
  float m  = fmaxf(fmaxf(a, b), fmaxf(c, d));
  float ea = __expf(a - m), eb = __expf(b - m), ec = __expf(c - m), ed = __expf(d - m);
  float inv = 1.0f / (ea + eb + ec + ed);
  p[0] = ea * inv; p[UDIM] = eb * inv; p[2 * UDIM] = ec * inv; p[3 * UDIM] = ed * inv;
}

// ---------------------------------------------------------------------------
// GEMM2: hc_pre = [rs | x] @ [u ; w_i].  Fused epilogue:
//   h = z1*L + z2*T + z3*D + z0*tanh(hc_pre + b_i)
// ---------------------------------------------------------------------------
__global__ __launch_bounds__(256) void gemm2_kernel(
    const bf16_t* __restrict__ A2, const bf16_t* __restrict__ W2b,
    const float* __restrict__ b_i, const float* __restrict__ gz,
    const float* __restrict__ left, const float* __restrict__ top,
    const float* __restrict__ diag, float* __restrict__ out)
{
  __shared__ bf16_t As[2 * ABUF];
  __shared__ bf16_t Bs[2 * BBUF];

  const int tilesN  = UDIM / TN;                  // 4
  const int rowTile = (blockIdx.x / tilesN) * TM;
  const int colTile = (blockIdx.x % tilesN) * TN;

  v8f acc[4][4];
  gemm_mainloop(A2, W2b, UDIM, rowTile, colTile, As, Bs, acc);

  const int lane  = threadIdx.x & 31;
  const int wave  = threadIdx.x >> 5;
  const int mBase = (wave >> 2) * 64;
  const int nBase = (wave & 3) * 64;

  #pragma unroll
  for (int i = 0; i < 4; ++i) {
    #pragma unroll
    for (int j = 0; j < 4; ++j) {
      const int col = colTile + nBase + 16 * j + (lane & 15);
      #pragma unroll
      for (int r = 0; r < 8; ++r) {
        const int row = rowTile + mBase + 16 * i + r + 8 * (lane >> 4);
        const float hc = tanhf(acc[i][j][r] + b_i[col]);
        const float* zr = gz + (size_t)row * DIMK;
        const float z0 = zr[col], z1 = zr[col + UDIM];
        const float z2 = zr[col + 2 * UDIM], z3 = zr[col + 3 * UDIM];
        const size_t s = (size_t)row * UDIM + col;
        out[s] = z1 * left[s] + z2 * top[s] + z3 * diag[s] + z0 * hc;
      }
    }
  }
}

// ---------------------------------------------------------------------------
// Pack kernels: fp32 -> bf16 operand assembly
// ---------------------------------------------------------------------------
__global__ __launch_bounds__(256) void pack_q_kernel(
    const float* __restrict__ x, const float* __restrict__ left,
    const float* __restrict__ top, const float* __restrict__ diag,
    bf16_t* __restrict__ q, bf16_t* __restrict__ A2)
{
  size_t t = (size_t)blockIdx.x * blockDim.x + threadIdx.x;
  if (t >= (size_t)BQ * DIMK) return;
  size_t row = t >> 12;
  int c = (int)(t & (DIMK - 1));
  float v;
  if (c < UDIM)            v = left[row * UDIM + c];
  else if (c < 2 * UDIM)   v = top [row * UDIM + (c - UDIM)];
  else if (c < 3 * UDIM)   v = diag[row * UDIM + (c - 2 * UDIM)];
  else { v = x[row * UDIM + (c - 3 * UDIM)]; A2[t] = f2bf(v); }  // x part of GEMM2 A
  q[t] = f2bf(v);
}

__global__ __launch_bounds__(256) void pack_w_kernel(
    const float* __restrict__ w0, const float* __restrict__ w1,
    const float* __restrict__ w2, const float* __restrict__ w3,
    const float* __restrict__ w4, const float* __restrict__ w5,
    const float* __restrict__ w6, bf16_t* __restrict__ Wb)
{
  size_t t = (size_t)blockIdx.x * blockDim.x + threadIdx.x;
  if (t >= (size_t)DIMK * N1) return;
  size_t k = t / N1;
  int n = (int)(t - k * N1);
  int gate = n >> 10, c = n & (UDIM - 1);
  const float* w = (gate == 0) ? w0 : (gate == 1) ? w1 : (gate == 2) ? w2 :
                   (gate == 3) ? w3 : (gate == 4) ? w4 : (gate == 5) ? w5 : w6;
  Wb[t] = f2bf(w[k * UDIM + c]);
}

__global__ __launch_bounds__(256) void pack_w2_kernel(
    const float* __restrict__ u, const float* __restrict__ wi,
    bf16_t* __restrict__ W2b)
{
  size_t t = (size_t)blockIdx.x * blockDim.x + threadIdx.x;
  if (t >= (size_t)DIMK * UDIM) return;
  size_t k = t >> 10;
  int n = (int)(t & (UDIM - 1));
  float v = (k < 3 * UDIM) ? u[k * UDIM + n] : wi[(k - 3 * UDIM) * UDIM + n];
  W2b[t] = f2bf(v);
}

__global__ __launch_bounds__(256) void pack_bias_kernel(
    const float* __restrict__ b0, const float* __restrict__ b1,
    const float* __restrict__ b2, const float* __restrict__ b3,
    const float* __restrict__ b4, const float* __restrict__ b5,
    const float* __restrict__ b6, float* __restrict__ bias)
{
  int t = blockIdx.x * blockDim.x + threadIdx.x;
  if (t >= N1) return;
  int gate = t >> 10, c = t & (UDIM - 1);
  const float* b = (gate == 0) ? b0 : (gate == 1) ? b1 : (gate == 2) ? b2 :
                   (gate == 3) ? b3 : (gate == 4) ? b4 : (gate == 5) ? b5 : b6;
  bias[t] = b[c];
}

// ---------------------------------------------------------------------------
extern "C" void kernel_launch(void* const* d_in, const int* in_sizes, int n_in,
                              void* d_out, int out_size, void* d_ws, size_t ws_size,
                              hipStream_t stream) {
  (void)in_sizes; (void)n_in; (void)out_size; (void)ws_size;
  const float* x    = (const float*)d_in[0];
  const float* left = (const float*)d_in[1];
  const float* top  = (const float*)d_in[2];
  const float* diag = (const float*)d_in[3];
  const float* w_rl = (const float*)d_in[4];   const float* b_rl = (const float*)d_in[5];
  const float* w_rt = (const float*)d_in[6];   const float* b_rt = (const float*)d_in[7];
  const float* w_rd = (const float*)d_in[8];   const float* b_rd = (const float*)d_in[9];
  const float* w_zi = (const float*)d_in[10];  const float* b_zi = (const float*)d_in[11];
  const float* w_zl = (const float*)d_in[12];  const float* b_zl = (const float*)d_in[13];
  const float* w_zt = (const float*)d_in[14];  const float* b_zt = (const float*)d_in[15];
  const float* w_zd = (const float*)d_in[16];  const float* b_zd = (const float*)d_in[17];
  const float* u    = (const float*)d_in[18];
  const float* w_i  = (const float*)d_in[19];
  const float* b_i  = (const float*)d_in[20];
  float* out = (float*)d_out;

  // workspace carve-up (256B aligned)
  char* ws = (char*)d_ws;
  size_t off = 0;
  auto carve = [&](size_t bytes) { char* p = ws + off; off += (bytes + 255) & ~(size_t)255; return p; };
  bf16_t* Wb   = (bf16_t*)carve((size_t)DIMK * N1 * sizeof(bf16_t));     //  58.7 MB
  bf16_t* qb   = (bf16_t*)carve((size_t)BQ * DIMK * sizeof(bf16_t));     //  67.1 MB
  bf16_t* A2   = (bf16_t*)carve((size_t)BQ * DIMK * sizeof(bf16_t));     //  67.1 MB
  float*  gz   = (float*) carve((size_t)BQ * DIMK * sizeof(float));      // 134.2 MB
  bf16_t* W2b  = (bf16_t*)carve((size_t)DIMK * UDIM * sizeof(bf16_t));   //   8.4 MB
  float*  bias = (float*) carve((size_t)N1 * sizeof(float));             //  28 KB

  const int T = 256;
  pack_bias_kernel<<<(N1 + T - 1) / T, T, 0, stream>>>(b_rl, b_rt, b_rd, b_zi, b_zl, b_zt, b_zd, bias);
  pack_w_kernel  <<<(unsigned)(((size_t)DIMK * N1 + T - 1) / T), T, 0, stream>>>(
      w_rl, w_rt, w_rd, w_zi, w_zl, w_zt, w_zd, Wb);
  pack_w2_kernel <<<(unsigned)(((size_t)DIMK * UDIM + T - 1) / T), T, 0, stream>>>(u, w_i, W2b);
  pack_q_kernel  <<<(unsigned)(((size_t)BQ * DIMK + T - 1) / T), T, 0, stream>>>(
      x, left, top, diag, qb, A2);

  gemm1_kernel<<<(BQ / TM) * (N1 / TN), T, 0, stream>>>(
      qb, Wb, bias, left, top, diag, A2, gz);

  softmax4_kernel<<<(unsigned)(((size_t)BQ * UDIM + T - 1) / T), T, 0, stream>>>(gz);

  gemm2_kernel<<<(BQ / TM) * (UDIM / TN), T, 0, stream>>>(
      A2, W2b, b_i, gz, left, top, diag, out);
}
